// RNNLayer_73426760893124
// MI455X (gfx1250) — matrix operand order repaired and checked by reading, not compiled, not run on hardware
//
#include <hip/hip_runtime.h>
#include <hip/hip_bf16.h>

// ---------------------------------------------------------------------------
// 2-layer tanh RNN for MI455X (gfx1250, wave32, WMMA).
//   h_t = tanh(x_t @ Wx^T + h_{t-1} @ Wh^T + b)
// Strategy:
//   * Input projections (no time dependency) -> big parallel bf16 WMMA GEMMs,
//     results stored PRE-SWIZZLED in WMMA C-fragment order so the scan's
//     per-step accumulator init is a single b128 load per tile.
//   * Recurrent scan -> single 32-wave workgroup; h kept in 64KB LDS (bf16),
//     Wh (512KB bf16) stays resident in L2; f32 accumulate via
//     v_wmma_f32_16x16x32_bf16; s_barrier between steps.
//   * Streaming fp32 output + read-once fp32 input use NT cache policy so the
//     L2-resident working set (Wh, Xproj, S0 ~ 135MB < 192MB) isn't evicted.
// ---------------------------------------------------------------------------

typedef __bf16 bf16_t;
typedef __bf16 v16bf __attribute__((ext_vector_type(16)));
typedef float  v8f   __attribute__((ext_vector_type(8)));
typedef float  f2v   __attribute__((ext_vector_type(2)));

#define T_STEPS 1024
#define BATCH   64
#define VDIM    256
#define HIDDEN  512

union ABPack {
  v16bf v;
  unsigned int u[8];
};

union CPack {          // 8 bf16 = one C-fragment column per lane (16 bytes)
  uint4  q;
  bf16_t h[8];
};

__device__ __forceinline__ v8f wmma_bf16(const ABPack& a, const ABPack& b, v8f c) {
  // 8 args: (neg_a, A, neg_b, B, c_mod, C, reuse_a, reuse_b)
  return __builtin_amdgcn_wmma_f32_16x16x32_bf16(false, a.v, false, b.v,
                                                 (short)0, c, false, false);
}

// Swizzled Xproj element index: one uint4 per (t, ntile, mtile, lane).
__device__ __forceinline__ size_t xsw_idx(int t, int nt, int mt, int lane) {
  return ((((size_t)t * (HIDDEN / 16) + nt) * 4 + mt) * 32 + lane);
}

// ---------------------------------------------------------------------------
// Convert fp32 weights into packed bf16 (row-major N x K), split into Wx / Wh.
// ---------------------------------------------------------------------------
__global__ void prep_weights(const float* __restrict__ Wnet,
                             const float* __restrict__ Wdeep,
                             bf16_t* __restrict__ Wx0, bf16_t* __restrict__ Wh0,
                             bf16_t* __restrict__ Wx1, bf16_t* __restrict__ Wh1) {
  const int n1 = HIDDEN * (VDIM + HIDDEN);   // 512*768
  const int n2 = HIDDEN * (2 * HIDDEN);      // 512*1024
  for (int i = blockIdx.x * blockDim.x + threadIdx.x; i < n1 + n2;
       i += gridDim.x * blockDim.x) {
    if (i < n1) {
      int n = i / (VDIM + HIDDEN), c = i % (VDIM + HIDDEN);
      float val = Wnet[i];
      if (c < VDIM) Wx0[n * VDIM + c]            = (bf16_t)val;
      else          Wh0[n * HIDDEN + (c - VDIM)] = (bf16_t)val;
    } else {
      int j = i - n1;
      int n = j / (2 * HIDDEN), c = j % (2 * HIDDEN);
      float val = Wdeep[j];
      if (c < HIDDEN) Wx1[n * HIDDEN + c]            = (bf16_t)val;
      else            Wh1[n * HIDDEN + (c - HIDDEN)] = (bf16_t)val;
    }
  }
}

// ---------------------------------------------------------------------------
// Xproj GEMM: Yswz <- swizzled bf16( X @ W^T + bias )
//   rows = T*B (65536, so one 64-row block == one timestep), K = 256 or 512.
// Each wave computes a 64x16 strip (4 M-tiles reuse one B-tile load).
// A layout per lane (16-bit A 16x32): VGPR v -> K = (v<4?0:16)+8*half+2*(v&3)
// B layout per lane (16-bit B 32x16): VGPR v -> K = (v<4?0:16)+8*half+2*(v&3), N = lane&15
// ---------------------------------------------------------------------------
template <bool IN_BF16>
__global__ void __launch_bounds__(512)
xproj_gemm(const void* __restrict__ Xin, const bf16_t* __restrict__ W,
           const float* __restrict__ bias, uint4* __restrict__ Yswz, int K) {
  const int lane = threadIdx.x & 31;
  const int gw   = blockIdx.x * (blockDim.x >> 5) + (threadIdx.x >> 5);
  const int NT   = HIDDEN / 16;         // 32 n-tiles
  const int mblk = gw / NT;             // 64-row block index == timestep t
  const int nt   = gw % NT;
  const int m    = lane & 15;
  const int half = lane >> 4;
  const int n    = nt * 16 + m;

  const bf16_t* Xb = (const bf16_t*)Xin;
  const float*  Xf = (const float*)Xin;

  v8f acc[4] = {};

  for (int k0 = 0; k0 < K; k0 += 32) {
    ABPack b;
#pragma unroll
    for (int v = 0; v < 8; ++v) {
      int kb = k0 + ((v < 4) ? 0 : 16) + 8 * half + 2 * (v & 3);
      b.u[v] = *(const unsigned int*)&W[(size_t)n * K + kb];
    }
#pragma unroll
    for (int mt = 0; mt < 4; ++mt) {
      ABPack a;
      const size_t row = (size_t)mblk * 64 + mt * 16 + m;
#pragma unroll
      for (int v = 0; v < 8; ++v) {
        int ka = k0 + ((v < 4) ? 0 : 16) + 8 * half + 2 * (v & 3);
        if (IN_BF16) {
          a.u[v] = *(const unsigned int*)&Xb[row * K + ka];
        } else {
          // read-once fp32 input stream: NT policy, don't pollute L2
          const f2v f = __builtin_nontemporal_load((const f2v*)&Xf[row * K + ka]);
          a.v[2 * v]     = (bf16_t)f.x;
          a.v[2 * v + 1] = (bf16_t)f.y;
        }
      }
      acc[mt] = wmma_bf16(a, b, acc[mt]);
    }
  }

  const float bn = bias[n];
#pragma unroll
  for (int mt = 0; mt < 4; ++mt) {
    CPack c;
#pragma unroll
    for (int r = 0; r < 8; ++r) c.h[r] = (bf16_t)(acc[mt][r] + bn);
    Yswz[xsw_idx(mblk, nt, mt, lane)] = c.q;   // RT: re-read by the scan via L2
  }
}

// ---------------------------------------------------------------------------
// Recurrent scan: single workgroup, 32 waves (1024 threads) on one WGP.
// h (64x512 bf16 = 64KB) lives in LDS. Wave w owns N-columns [16w,16w+16) for
// all 64 batch rows (4 M-tiles). Per step: 16 k-steps x 4 tiles = 64 WMMAs
// per wave (2048 for the step). Read-phase / write-phase split by barriers.
// Accumulator init: 4x global_load_b128 from the swizzled Xproj.
// ---------------------------------------------------------------------------
__global__ void __launch_bounds__(1024)
rnn_scan(const uint4*  __restrict__ Xswz,     // swizzled (T,32,4,32) bf16x8
         const bf16_t* __restrict__ Wh,       // (HID,HID) bf16 row-major
         const float*  __restrict__ h0,       // (B,HID) f32
         bf16_t* __restrict__ states_bf,      // (T,B,HID) bf16 or null
         float*  __restrict__ states_f32,     // (T,B,HID) f32 or null (NT)
         float*  __restrict__ last_out) {     // (B,HID) f32
  __shared__ bf16_t hsh[BATCH * HIDDEN];      // exactly 64 KB

  const int lane = threadIdx.x & 31;
  const int w    = threadIdx.x >> 5;          // wave id: 0..31 = n-tile
  const int m    = lane & 15;
  const int half = lane >> 4;
  const int n    = w * 16 + m;

  for (int i = threadIdx.x; i < BATCH * HIDDEN; i += 1024)
    hsh[i] = (bf16_t)h0[i];
  __syncthreads();

  for (int t = 0; t < T_STEPS; ++t) {
    // init accumulators from swizzled x-projection (+bias): 1 b128 per tile
    v8f acc[4];
#pragma unroll
    for (int mt = 0; mt < 4; ++mt) {
      CPack c;
      c.q = Xswz[xsw_idx(t, w, mt, lane)];
#pragma unroll
      for (int r = 0; r < 8; ++r) acc[mt][r] = (float)c.h[r];
    }

    // recurrent GEMM: h_{t-1} (LDS) x Wh^T (L2-resident)
#pragma unroll 4
    for (int ks = 0; ks < HIDDEN / 32; ++ks) {
      const int k0 = ks * 32;
      ABPack b;
#pragma unroll
      for (int v = 0; v < 8; ++v)
        b.u[v] = *(const unsigned int*)&Wh[(size_t)n * HIDDEN + k0 + ((v < 4) ? 0 : 16) + 8 * half + 2 * (v & 3)];
#pragma unroll
      for (int mt = 0; mt < 4; ++mt) {
        ABPack a;
#pragma unroll
        for (int v = 0; v < 8; ++v) {
          int ka = k0 + ((v < 4) ? 0 : 16) + 8 * half + 2 * (v & 3);
          a.u[v] = *(const unsigned int*)&hsh[(mt * 16 + m) * HIDDEN + ka];
        }
        acc[mt] = wmma_bf16(a, b, acc[mt]);
      }
    }

    __syncthreads();   // everyone done READING h_{t-1}

#pragma unroll
    for (int mt = 0; mt < 4; ++mt)
#pragma unroll
      for (int r = 0; r < 8; ++r) {
        const int   row = mt * 16 + r + 8 * half;   // batch index
        const float hv  = tanhf(acc[mt][r]);
        const bf16_t hb = (bf16_t)hv;
        hsh[row * HIDDEN + n] = hb;
        const size_t gi = ((size_t)t * BATCH + row) * HIDDEN + n;
        if (states_bf)  states_bf[gi] = hb;                       // RT: L2 reuse
        if (states_f32) __builtin_nontemporal_store(hv, &states_f32[gi]); // stream
        if (t == T_STEPS - 1) last_out[row * HIDDEN + n] = hv;
      }

    __syncthreads();   // h_t fully published before next step reads it
  }
}

// ---------------------------------------------------------------------------
extern "C" void kernel_launch(void* const* d_in, const int* in_sizes, int n_in,
                              void* d_out, int out_size, void* d_ws, size_t ws_size,
                              hipStream_t stream) {
  (void)in_sizes; (void)n_in; (void)out_size; (void)ws_size;

  const float* inputs = (const float*)d_in[0];   // (T,B,V)
  const float* Hin    = (const float*)d_in[1];   // (L,B,HID)
  const float* W_net  = (const float*)d_in[2];   // (HID, V+HID)
  const float* b_net  = (const float*)d_in[3];   // (HID)
  const float* W_deep = (const float*)d_in[4];   // (HID, 2*HID)
  const float* b_deep = (const float*)d_in[5];   // (HID)
  float* out = (float*)d_out;

  // workspace carve-up (~136 MB): bf16 weights + Xproj (reused) + layer0 states
  char* ws = (char*)d_ws;
  bf16_t* Wx0 = (bf16_t*)ws;  ws += (size_t)HIDDEN * VDIM   * sizeof(bf16_t);
  bf16_t* Wh0 = (bf16_t*)ws;  ws += (size_t)HIDDEN * HIDDEN * sizeof(bf16_t);
  bf16_t* Wx1 = (bf16_t*)ws;  ws += (size_t)HIDDEN * HIDDEN * sizeof(bf16_t);
  bf16_t* Wh1 = (bf16_t*)ws;  ws += (size_t)HIDDEN * HIDDEN * sizeof(bf16_t);
  uint4*  Xproj = (uint4*)ws; ws += (size_t)T_STEPS * BATCH * HIDDEN * sizeof(bf16_t);
  bf16_t* S0    = (bf16_t*)ws; ws += (size_t)T_STEPS * BATCH * HIDDEN * sizeof(bf16_t);

  float* out_states = out;                                        // (T,B,HID)
  float* out_last0  = out + (size_t)T_STEPS * BATCH * HIDDEN;     // (B,HID)
  float* out_last1  = out_last0 + (size_t)BATCH * HIDDEN;         // (B,HID)

  prep_weights<<<896, 256, 0, stream>>>(W_net, W_deep, Wx0, Wh0, Wx1, Wh1);

  const int waves  = (T_STEPS * BATCH / 64) * (HIDDEN / 16);  // 32768
  const int blocks = waves / 16;                              // 512 thr = 16 waves

  // layer 0: x-projection (fp32 inputs, K=256), then scan
  xproj_gemm<false><<<blocks, 512, 0, stream>>>(inputs, Wx0, b_net, Xproj, VDIM);
  rnn_scan<<<1, 1024, 0, stream>>>(Xproj, Wh0, Hin, S0, nullptr, out_last0);

  // layer 1: x-projection from bf16 layer-0 states (K=512), then scan
  xproj_gemm<true><<<blocks, 512, 0, stream>>>(S0, Wx1, b_deep, Xproj, HIDDEN);
  rnn_scan<<<1, 1024, 0, stream>>>(Xproj, Wh1, Hin + (size_t)BATCH * HIDDEN,
                                   nullptr, out_states, out_last1);
}
